// attention_78254304133707
// MI455X (gfx1250) — matrix operand order repaired and checked by reading, not compiled, not run on hardware
//
#include <hip/hip_runtime.h>

// ---------------- problem constants ----------------
#define BB 4
#define CC 256
#define NN 16384            // 128*128
#define EPSV 1e-5f
#define XT_STRIDE 264       // bf16 elements per LDS row: 528B = 33*16B (aligned, odd*4 dword bank stride)

typedef __attribute__((ext_vector_type(16))) __bf16        v16bf;
typedef __attribute__((ext_vector_type(8)))  float         v8f;
typedef __attribute__((ext_vector_type(4)))  unsigned int  v4u;
typedef __attribute__((ext_vector_type(8)))  unsigned int  v8u;

__device__ __forceinline__ unsigned short f2bf(float f) {
    unsigned u = __float_as_uint(f);
    u += 0x7FFFu + ((u >> 16) & 1u);       // round-to-nearest-even
    return (unsigned short)(u >> 16);
}

__device__ __forceinline__ v4u ld16(const unsigned short* p) {
    return *reinterpret_cast<const v4u*>(p);
}

__device__ __forceinline__ v16bf make_frag(v4u lo, v4u hi) {
    v8u u;
    u[0] = lo[0]; u[1] = lo[1]; u[2] = lo[2]; u[3] = lo[3];
    u[4] = hi[0]; u[5] = hi[1]; u[6] = hi[2]; u[7] = hi[3];
    return __builtin_bit_cast(v16bf, u);
}

__device__ __forceinline__ v8f wmma_bf16(v16bf a, v16bf b, v8f c) {
    return __builtin_amdgcn_wmma_f32_16x16x32_bf16(
        false, a, false, b, (short)0, c, false, false);
}

__device__ __forceinline__ v8f zero8() {
    v8f z = {0.f, 0.f, 0.f, 0.f, 0.f, 0.f, 0.f, 0.f};
    return z;
}

// ---------------- K0: per-(b,c) mean / rstd over spatial dims ----------------
__global__ __launch_bounds__(256) void stats_kernel(
    const float* __restrict__ content, const float* __restrict__ style,
    float* __restrict__ meanArr, float* __restrict__ rstdArr)
{
    int idx = blockIdx.x;                           // 0..2047 : [content rows | style rows]
    const float* src = (idx < BB * CC) ? content : style;
    int row = idx & (BB * CC - 1);
    const float* p = src + (size_t)row * NN;

    float s = 0.f, sq = 0.f;
    for (int i = threadIdx.x; i < NN; i += 256) {
        float v = p[i];
        s += v; sq += v * v;
    }
    __shared__ float ls[256], lq[256];
    ls[threadIdx.x] = s; lq[threadIdx.x] = sq;
    __syncthreads();
    for (int off = 128; off > 0; off >>= 1) {
        if (threadIdx.x < off) {
            ls[threadIdx.x] += ls[threadIdx.x + off];
            lq[threadIdx.x] += lq[threadIdx.x + off];
        }
        __syncthreads();
    }
    if (threadIdx.x == 0) {
        float m   = ls[0] * (1.f / NN);
        float var = lq[0] * (1.f / NN) - m * m;
        meanArr[idx] = m;
        rstdArr[idx] = rsqrtf(var + EPSV);
    }
}

// ---------------- K1: weights f32 -> bf16 (wq|wk|wv|wo packed) ----------------
__global__ __launch_bounds__(256) void cvt_w_kernel(
    const float* __restrict__ wq, const float* __restrict__ wk,
    const float* __restrict__ wv, const float* __restrict__ wo,
    unsigned short* __restrict__ out)
{
    int i = blockIdx.x * 256 + threadIdx.x;         // 0 .. 4*65536-1
    int which = i >> 16, j = i & 0xFFFF;
    const float* w = (which == 0) ? wq : (which == 1) ? wk : (which == 2) ? wv : wo;
    out[i] = f2bf(w[j]);
}

// ---------------- K2: Q/K/V projection GEMMs (M=256,K=256 per 64-wide n tile) --
__global__ __launch_bounds__(256) void proj_kernel(
    const float* __restrict__ content, const float* __restrict__ style,
    const float* __restrict__ meanArr, const float* __restrict__ rstdArr,
    const unsigned short* __restrict__ wbf,
    const float* __restrict__ bq, const float* __restrict__ bk, const float* __restrict__ bv,
    unsigned short* __restrict__ Qb, unsigned short* __restrict__ Kb,
    unsigned short* __restrict__ Vb)
{
    int z = blockIdx.z;                  // b*3 + which
    int b = z / 3, which = z % 3;
    int n0 = blockIdx.x * 64;

    const float* x   = (which == 0) ? content : style;
    const float* bia = (which == 0) ? bq : (which == 1) ? bk : bv;
    const unsigned short* w = wbf + (size_t)which * (CC * CC);
    unsigned short* out = (which == 0) ? Qb : (which == 1) ? Kb : Vb;
    int statBase = ((which == 0) ? 0 : BB * CC) + b * CC;
    bool doNorm = (which != 2);

    __shared__ __align__(16) unsigned short XT[64 * XT_STRIDE]; // [n][c] transposed tile
    {
        int tn = threadIdx.x & 63;
        int c0 = threadIdx.x >> 6;
        for (int c = c0; c < CC; c += 4) {
            float m = doNorm ? meanArr[statBase + c] : 0.f;
            float r = doNorm ? rstdArr[statBase + c] : 1.f;
            float v = x[((size_t)(b * CC + c)) * NN + n0 + tn];
            XT[tn * XT_STRIDE + c] = f2bf((v - m) * r);
        }
    }
    __syncthreads();

    int lane = threadIdx.x & 31, wid = threadIdx.x >> 5;
    int mW = wid * 32;                   // 8 waves cover M=256
    int hi = (lane >> 4) & 1, ln = lane & 15;

    v8f acc[2][4];
#pragma unroll
    for (int mt = 0; mt < 2; mt++)
#pragma unroll
        for (int nt = 0; nt < 4; nt++) acc[mt][nt] = zero8();

    for (int k0 = 0; k0 < CC; k0 += 32) {
        v16bf a[2], bf[4];
#pragma unroll
        for (int mt = 0; mt < 2; mt++) {
            const unsigned short* ap = w + (size_t)(mW + mt * 16 + ln) * CC + k0 + hi * 8;
            a[mt] = make_frag(ld16(ap), ld16(ap + 16));
        }
#pragma unroll
        for (int nt = 0; nt < 4; nt++) {
            const unsigned short* bp = &XT[(nt * 16 + ln) * XT_STRIDE + k0 + hi * 8];
            bf[nt] = make_frag(ld16(bp), ld16(bp + 16));
        }
#pragma unroll
        for (int mt = 0; mt < 2; mt++)
#pragma unroll
            for (int nt = 0; nt < 4; nt++)
                acc[mt][nt] = wmma_bf16(a[mt], bf[nt], acc[mt][nt]);
    }

#pragma unroll
    for (int mt = 0; mt < 2; mt++)
#pragma unroll
        for (int nt = 0; nt < 4; nt++)
#pragma unroll
            for (int r = 0; r < 8; r++) {
                int m = mW + mt * 16 + r + hi * 8;
                int n = n0 + nt * 16 + ln;
                out[((size_t)(b * CC + m)) * NN + n] = f2bf(acc[mt][nt][r] + bia[m]);
            }
}

// ---------------- K3: scores = Q K^T  (K-dim = N = 16384) --------------------
__global__ __launch_bounds__(256) void scores_kernel(
    const unsigned short* __restrict__ Qb, const unsigned short* __restrict__ Kb,
    float* __restrict__ scores)
{
    int b  = blockIdx.z;
    int m0 = blockIdx.y * 64;
    int n0 = blockIdx.x * 128;
    int lane = threadIdx.x & 31, wid = threadIdx.x >> 5;
    int wm = m0 + (wid >> 2) * 32;
    int wn = n0 + (wid & 3) * 32;
    int hi = (lane >> 4) & 1, ln = lane & 15;

    const unsigned short* Qp = Qb + (size_t)b * CC * NN;
    const unsigned short* Kp = Kb + (size_t)b * CC * NN;

    v8f acc[2][2];
#pragma unroll
    for (int mt = 0; mt < 2; mt++)
#pragma unroll
        for (int nt = 0; nt < 2; nt++) acc[mt][nt] = zero8();

    for (int k0 = 0; k0 < NN; k0 += 32) {
        v16bf a[2], bf[2];
#pragma unroll
        for (int mt = 0; mt < 2; mt++) {
            const unsigned short* ap = Qp + (size_t)(wm + mt * 16 + ln) * NN + k0 + hi * 8;
            __builtin_prefetch(ap + 512, 0, 1);   // global_prefetch_b8 ahead on K
            a[mt] = make_frag(ld16(ap), ld16(ap + 16));
        }
#pragma unroll
        for (int nt = 0; nt < 2; nt++) {
            const unsigned short* bp = Kp + (size_t)(wn + nt * 16 + ln) * NN + k0 + hi * 8;
            __builtin_prefetch(bp + 512, 0, 1);
            bf[nt] = make_frag(ld16(bp), ld16(bp + 16));
        }
#pragma unroll
        for (int mt = 0; mt < 2; mt++)
#pragma unroll
            for (int nt = 0; nt < 2; nt++)
                acc[mt][nt] = wmma_bf16(a[mt], bf[nt], acc[mt][nt]);
    }

    float* sp = scores + (size_t)b * CC * CC;
#pragma unroll
    for (int mt = 0; mt < 2; mt++)
#pragma unroll
        for (int nt = 0; nt < 2; nt++)
#pragma unroll
            for (int r = 0; r < 8; r++) {
                int m = wm + mt * 16 + r + hi * 8;
                int n = wn + nt * 16 + ln;
                sp[(size_t)m * CC + n] = acc[mt][nt][r];
            }
}

// ---------------- K4: row softmax -> attn (bf16) -----------------------------
__global__ __launch_bounds__(256) void softmax_kernel(
    const float* __restrict__ scores, unsigned short* __restrict__ attn)
{
    int row = blockIdx.x;                          // b*C + c  (0..1023)
    const float* s = scores + (size_t)row * CC;
    float v = s[threadIdx.x];

    __shared__ float red[256];
    red[threadIdx.x] = v;
    __syncthreads();
    for (int off = 128; off > 0; off >>= 1) {
        if (threadIdx.x < off)
            red[threadIdx.x] = fmaxf(red[threadIdx.x], red[threadIdx.x + off]);
        __syncthreads();
    }
    float mx = red[0];
    __syncthreads();

    float e = __expf(v - mx);
    red[threadIdx.x] = e;
    __syncthreads();
    for (int off = 128; off > 0; off >>= 1) {
        if (threadIdx.x < off)
            red[threadIdx.x] += red[threadIdx.x + off];
        __syncthreads();
    }
    float inv = 1.f / red[0];
    attn[(size_t)row * CC + threadIdx.x] = f2bf(e * inv);
}

// ---------------- K5: astyle = attn*V ; out = wo*astyle + bo + content -------
__global__ __launch_bounds__(256) void out_kernel(
    const unsigned short* __restrict__ attn, const unsigned short* __restrict__ Vb,
    const unsigned short* __restrict__ wo_bf, const float* __restrict__ bo,
    const float* __restrict__ content, float* __restrict__ out)
{
    int b  = blockIdx.z;
    int n0 = blockIdx.x * 64;

    __shared__ __align__(16) unsigned short ST[64 * XT_STRIDE]; // V^T, then astyle^T

    int lane = threadIdx.x & 31, wid = threadIdx.x >> 5;
    int mW = wid * 32;
    int hi = (lane >> 4) & 1, ln = lane & 15;

    // ---- stage V tile transposed: ST[n][c] ----
    {
        int tn = threadIdx.x & 63, c0 = threadIdx.x >> 6;
        const unsigned short* Vp = Vb + (size_t)b * CC * NN;
        for (int c = c0; c < CC; c += 4)
            ST[tn * XT_STRIDE + c] = Vp[(size_t)c * NN + n0 + tn];
    }
    __syncthreads();

    // ---- GEMM 1: astyle[:, tile] = attn (CxC) * V[:, tile], K = C ----
    const unsigned short* Ap = attn + (size_t)b * CC * CC;
    v8f acc[2][4];
#pragma unroll
    for (int mt = 0; mt < 2; mt++)
#pragma unroll
        for (int nt = 0; nt < 4; nt++) acc[mt][nt] = zero8();

    for (int k0 = 0; k0 < CC; k0 += 32) {
        v16bf a[2], bf[4];
#pragma unroll
        for (int mt = 0; mt < 2; mt++) {
            const unsigned short* ap = Ap + (size_t)(mW + mt * 16 + ln) * CC + k0 + hi * 8;
            a[mt] = make_frag(ld16(ap), ld16(ap + 16));
        }
#pragma unroll
        for (int nt = 0; nt < 4; nt++) {
            const unsigned short* bp = &ST[(nt * 16 + ln) * XT_STRIDE + k0 + hi * 8];
            bf[nt] = make_frag(ld16(bp), ld16(bp + 16));
        }
#pragma unroll
        for (int mt = 0; mt < 2; mt++)
#pragma unroll
            for (int nt = 0; nt < 4; nt++)
                acc[mt][nt] = wmma_bf16(a[mt], bf[nt], acc[mt][nt]);
    }

    // ---- write astyle^T into the same LDS buffer ----
    __syncthreads();            // everyone done reading V^T
#pragma unroll
    for (int mt = 0; mt < 2; mt++)
#pragma unroll
        for (int nt = 0; nt < 4; nt++)
#pragma unroll
            for (int r = 0; r < 8; r++) {
                int m = mW + mt * 16 + r + hi * 8;
                int nl = nt * 16 + ln;
                ST[nl * XT_STRIDE + m] = f2bf(acc[mt][nt][r]);
            }
    __syncthreads();

    // ---- GEMM 2: out[:, tile] = wo (CxC) * astyle[:, tile], K = C ----
    v8f acc2[2][4];
#pragma unroll
    for (int mt = 0; mt < 2; mt++)
#pragma unroll
        for (int nt = 0; nt < 4; nt++) acc2[mt][nt] = zero8();

    for (int k0 = 0; k0 < CC; k0 += 32) {
        v16bf a[2], bf[4];
#pragma unroll
        for (int mt = 0; mt < 2; mt++) {
            const unsigned short* ap = wo_bf + (size_t)(mW + mt * 16 + ln) * CC + k0 + hi * 8;
            a[mt] = make_frag(ld16(ap), ld16(ap + 16));
        }
#pragma unroll
        for (int nt = 0; nt < 4; nt++) {
            const unsigned short* bp = &ST[(nt * 16 + ln) * XT_STRIDE + k0 + hi * 8];
            bf[nt] = make_frag(ld16(bp), ld16(bp + 16));
        }
#pragma unroll
        for (int mt = 0; mt < 2; mt++)
#pragma unroll
            for (int nt = 0; nt < 4; nt++)
                acc2[mt][nt] = wmma_bf16(a[mt], bf[nt], acc2[mt][nt]);
    }

    // ---- epilogue: + bo + content, f32 store ----
#pragma unroll
    for (int mt = 0; mt < 2; mt++)
#pragma unroll
        for (int nt = 0; nt < 4; nt++)
#pragma unroll
            for (int r = 0; r < 8; r++) {
                int o = mW + mt * 16 + r + hi * 8;
                int n = n0 + nt * 16 + ln;
                size_t idx = ((size_t)(b * CC + o)) * NN + n;
                out[idx] = acc2[mt][nt][r] + bo[o] + content[idx];
            }
}

// ---------------- host launcher ----------------------------------------------
extern "C" void kernel_launch(void* const* d_in, const int* in_sizes, int n_in,
                              void* d_out, int out_size, void* d_ws, size_t ws_size,
                              hipStream_t stream)
{
    const float* content = (const float*)d_in[0];
    const float* style   = (const float*)d_in[1];
    const float* wq = (const float*)d_in[2];
    const float* bq = (const float*)d_in[3];
    const float* wk = (const float*)d_in[4];
    const float* bk = (const float*)d_in[5];
    const float* wv = (const float*)d_in[6];
    const float* bv = (const float*)d_in[7];
    const float* wo = (const float*)d_in[8];
    const float* bo = (const float*)d_in[9];
    float* outp = (float*)d_out;

    char* ws = (char*)d_ws;
    const size_t nBCN = (size_t)BB * CC * NN;              // 16,777,216 elements
    float*          meanArr = (float*)(ws);                // 2048 f32
    float*          rstdArr = meanArr + 2 * BB * CC;       // 2048 f32
    unsigned short* wbf     = (unsigned short*)(ws + 16384);          // 4*C*C bf16
    unsigned short* Qb      = (unsigned short*)(ws + 16384 + 4 * CC * CC * 2);
    unsigned short* Kb      = Qb + nBCN;
    unsigned short* Vb      = Kb + nBCN;
    float*          scores  = (float*)(Vb + nBCN);                    // B*C*C f32
    unsigned short* attn    = (unsigned short*)(scores + (size_t)BB * CC * CC);

    unsigned short* wo_bf = wbf + (size_t)3 * CC * CC;

    stats_kernel<<<dim3(2 * BB * CC), dim3(256), 0, stream>>>(content, style, meanArr, rstdArr);
    cvt_w_kernel<<<dim3(4 * CC * CC / 256), dim3(256), 0, stream>>>(wq, wk, wv, wo, wbf);
    proj_kernel<<<dim3(NN / 64, 1, 3 * BB), dim3(256), 0, stream>>>(
        content, style, meanArr, rstdArr, wbf, bq, bk, bv, Qb, Kb, Vb);
    scores_kernel<<<dim3(CC / 128, CC / 64, BB), dim3(256), 0, stream>>>(Qb, Kb, scores);
    softmax_kernel<<<dim3(BB * CC), dim3(256), 0, stream>>>(scores, attn);
    out_kernel<<<dim3(NN / 64, 1, BB), dim3(256), 0, stream>>>(
        attn, Vb, wo_bf, bo, content, outp);
}
